// EnhancedPointNet2_11003706212680
// MI455X (gfx1250) — matrix operand order repaired and checked by reading, not compiled, not run on hardware
//
#include <hip/hip_runtime.h>
#include <hip/hip_bf16.h>
#include <math.h>

// ---------------------------------------------------------------------------
// Types for CDNA5 WMMA
// ---------------------------------------------------------------------------
typedef __attribute__((ext_vector_type(16))) _Float16 v16h;
typedef __attribute__((ext_vector_type(4)))  _Float16 v4h;
typedef __attribute__((ext_vector_type(8)))  float    v8f;

#define TM 128  // rows per block (8 waves x 16)
#define TN 64   // cols per block (4 x 16 per wave)
#define TK 32   // k-slice per WMMA

// ---------------------------------------------------------------------------
// Generic GEMM:  Y[M,N] = X[M,K] @ W[N,K]^T + bias[N]   (fp32 in/out,
// f16 staged through LDS, f32 accumulation via v_wmma_f32_16x16x32_f16).
// 256 threads = 8 waves; wave w computes rows [w*16,w*16+16) x all 64 cols
// (4 WMMA tiles, A-fragment reused across the 4 B fragments).
// Fast staging path: float4 global loads + v4h LDS stores when the tile is
// full and rows are 16B-aligned (K%4==0); guarded scalar path otherwise.
// ---------------------------------------------------------------------------
__global__ __launch_bounds__(256)
void gemm_bias_f16wmma(const float* __restrict__ X, const float* __restrict__ W,
                       const float* __restrict__ bias, float* __restrict__ Y,
                       int M, int N, int K)
{
    __shared__ _Float16 As[TM][TK];   // 8 KB
    __shared__ _Float16 Bs[TN][TK];   // 4 KB
    const int wave = threadIdx.x >> 5;
    const int lane = threadIdx.x & 31;
    const int m_base = blockIdx.y * TM;
    const int n_base = blockIdx.x * TN;
    v8f acc0 = {}, acc1 = {}, acc2 = {}, acc3 = {};

    const bool vec_ok = ((K & 3) == 0);
    const bool m_full = (m_base + TM <= M);
    const bool n_full = (n_base + TN <= N);

    for (int k0 = 0; k0 < K; k0 += TK) {
        const bool k_full = (k0 + TK <= K);

        if (vec_ok && m_full && k_full) {
            // 128 rows x 8 float4 = 1024 vec loads, 4 per thread
#pragma unroll
            for (int v = 0; v < 4; ++v) {
                int idx = (int)threadIdx.x + v * 256;
                int r = idx >> 3, c4 = (idx & 7) << 2;
                const float4 f = *(const float4*)&X[(size_t)(m_base + r) * K + k0 + c4];
                v4h h = { (_Float16)f.x, (_Float16)f.y, (_Float16)f.z, (_Float16)f.w };
                *(v4h*)&As[r][c4] = h;
            }
        } else {
            for (int i = threadIdx.x; i < TM * TK; i += 256) {
                int r = i >> 5, c = i & 31;
                int gm = m_base + r, gk = k0 + c;
                As[r][c] = (_Float16)((gm < M && gk < K) ? X[(size_t)gm * K + gk] : 0.0f);
            }
        }
        if (vec_ok && n_full && k_full) {
            // 64 rows x 8 float4 = 512 vec loads, 2 per thread
#pragma unroll
            for (int v = 0; v < 2; ++v) {
                int idx = (int)threadIdx.x + v * 256;
                int r = idx >> 3, c4 = (idx & 7) << 2;
                const float4 f = *(const float4*)&W[(size_t)(n_base + r) * K + k0 + c4];
                v4h h = { (_Float16)f.x, (_Float16)f.y, (_Float16)f.z, (_Float16)f.w };
                *(v4h*)&Bs[r][c4] = h;
            }
        } else {
            for (int i = threadIdx.x; i < TN * TK; i += 256) {
                int r = i >> 5, c = i & 31;
                int gn = n_base + r, gk = k0 + c;
                Bs[r][c] = (_Float16)((gn < N && gk < K) ? W[(size_t)gn * K + gk] : 0.0f);
            }
        }
        if (k0 + TK < K && m_full) {
            int gm = m_base + (threadIdx.x >> 1);
            __builtin_prefetch(&X[(size_t)gm * K + k0 + TK], 0, 1);
        }
        __syncthreads();

        // Fragments in contiguous k order (k = 16*(lane/16) + i), same for A(16x32) and B(32x16).
        v16h a, b0, b1, b2, b3;
        const int mrow = (wave << 4) + (lane & 15);
        const int ncol = lane & 15;
        const int ka   = (lane >> 4) << 4;   // A: upper half-wave holds k+16
        const int kb   = (lane >> 4) << 4;   // B: upper half-wave holds k+16
#pragma unroll
        for (int i = 0; i < 16; ++i) {
            a[i]  = As[mrow][i + ka];
            b0[i] = Bs[ncol +  0][i + kb];
            b1[i] = Bs[ncol + 16][i + kb];
            b2[i] = Bs[ncol + 32][i + kb];
            b3[i] = Bs[ncol + 48][i + kb];
        }
        acc0 = __builtin_amdgcn_wmma_f32_16x16x32_f16(false, a, false, b0, (short)0, acc0, false, false);
        acc1 = __builtin_amdgcn_wmma_f32_16x16x32_f16(false, a, false, b1, (short)0, acc1, false, false);
        acc2 = __builtin_amdgcn_wmma_f32_16x16x32_f16(false, a, false, b2, (short)0, acc2, false, false);
        acc3 = __builtin_amdgcn_wmma_f32_16x16x32_f16(false, a, false, b3, (short)0, acc3, false, false);
        __syncthreads();
    }

    // C/D layout: vgpr j -> row = wave*16 + (lane>=16 ? 8:0) + j, col = lane&15
    const int mrow_o = m_base + (wave << 4) + ((lane >> 4) << 3);
#pragma unroll
    for (int t = 0; t < 4; ++t) {
        v8f acc = (t == 0) ? acc0 : (t == 1) ? acc1 : (t == 2) ? acc2 : acc3;
        const int n = n_base + t * 16 + (lane & 15);
        if (n < N) {
            const float bv = bias ? bias[n] : 0.0f;
#pragma unroll
            for (int j = 0; j < 8; ++j) {
                const int m = mrow_o + j;
                if (m < M) Y[(size_t)m * N + n] = acc[j] + bv;
            }
        }
    }
}

// ---------------------------------------------------------------------------
// BatchNorm: per-channel mean/rstd over all M rows (fp32, two kernels)
// ---------------------------------------------------------------------------
__global__ void bn_stats(const float* __restrict__ Y, int M, int C,
                         float* __restrict__ mean, float* __restrict__ rstd)
{
    const int c = blockIdx.x;
    float s = 0.f, s2 = 0.f;
    for (int m = threadIdx.x; m < M; m += blockDim.x) {
        float v = Y[(size_t)m * C + c];
        s += v; s2 += v * v;
    }
    __shared__ float sh[256], sh2[256];
    sh[threadIdx.x] = s; sh2[threadIdx.x] = s2;
    __syncthreads();
    for (int o = blockDim.x >> 1; o > 0; o >>= 1) {
        if ((int)threadIdx.x < o) {
            sh[threadIdx.x]  += sh[threadIdx.x + o];
            sh2[threadIdx.x] += sh2[threadIdx.x + o];
        }
        __syncthreads();
    }
    if (threadIdx.x == 0) {
        float mu  = sh[0] / (float)M;
        float var = sh2[0] / (float)M - mu * mu;
        mean[c] = mu;
        rstd[c] = rsqrtf(fmaxf(var, 0.f) + 1e-5f);
    }
}

// y = relu((y-mean)*rstd*g + be) [+ residual], written (possibly strided) to dst
__global__ void bn_apply_relu(const float* __restrict__ Y, int M, int C,
                              const float* __restrict__ mean, const float* __restrict__ rstd,
                              const float* __restrict__ g, const float* __restrict__ be,
                              const float* __restrict__ residual,
                              float* __restrict__ dst, int ldd, int dcol)
{
    size_t t = (size_t)blockIdx.x * blockDim.x + threadIdx.x;
    size_t total = (size_t)M * C;
    if (t >= total) return;
    int c = (int)(t % C);
    size_t m = t / C;
    float v = Y[t];
    v = (v - mean[c]) * rstd[c] * g[c] + be[c];
    v = fmaxf(v, 0.f);
    if (residual) v += residual[m * (size_t)C + c];
    dst[m * (size_t)ldd + dcol + c] = v;
}

// ---------------------------------------------------------------------------
// Max-pool over the K (group) dimension: [Rows,K,C] -> dst[Rows, ldd] @ dcol
// ---------------------------------------------------------------------------
__global__ void maxpool_over_k(const float* __restrict__ src, int K, int C, int Rows,
                               float* __restrict__ dst, int ldd, int dcol)
{
    size_t t = (size_t)blockIdx.x * blockDim.x + threadIdx.x;
    size_t total = (size_t)Rows * C;
    if (t >= total) return;
    int c = (int)(t % C);
    size_t r = t / C;
    float mx = -3.4e38f;
    for (int kk = 0; kk < K; ++kk)
        mx = fmaxf(mx, src[((r * K) + kk) * (size_t)C + c]);
    dst[r * (size_t)ldd + dcol + c] = mx;
}

// ---------------------------------------------------------------------------
// Farthest point sampling: one block per batch, iterative argmax
// ---------------------------------------------------------------------------
__global__ void fps_kernel(const float* __restrict__ xyz, int B_, int N, int S,
                           int* __restrict__ out_idx, float* __restrict__ dist_ws)
{
    const int b = blockIdx.x;
    const float* p = xyz + (size_t)b * N * 3;
    float* dist = dist_ws + (size_t)b * N;
    for (int i = threadIdx.x; i < N; i += blockDim.x) dist[i] = 1e10f;
    __shared__ int s_far;
    __shared__ float red_v[256];
    __shared__ int   red_i[256];
    if (threadIdx.x == 0) s_far = 0;
    __syncthreads();
    for (int it = 0; it < S; ++it) {
        const int far = s_far;
        if (threadIdx.x == 0) out_idx[(size_t)b * S + it] = far;
        const float cx = p[far * 3], cy = p[far * 3 + 1], cz = p[far * 3 + 2];
        float best = -1.f; int bi = 0;
        for (int i = threadIdx.x; i < N; i += blockDim.x) {
            float dx = p[i * 3] - cx, dy = p[i * 3 + 1] - cy, dz = p[i * 3 + 2] - cz;
            float d  = dx * dx + dy * dy + dz * dz;
            float dm = fminf(dist[i], d);
            dist[i] = dm;
            if (dm > best) { best = dm; bi = i; }
        }
        red_v[threadIdx.x] = best; red_i[threadIdx.x] = bi;
        __syncthreads();
        for (int o = blockDim.x >> 1; o > 0; o >>= 1) {
            if ((int)threadIdx.x < o && red_v[threadIdx.x + o] > red_v[threadIdx.x]) {
                red_v[threadIdx.x] = red_v[threadIdx.x + o];
                red_i[threadIdx.x] = red_i[threadIdx.x + o];
            }
            __syncthreads();
        }
        if (threadIdx.x == 0) s_far = red_i[0];
        __syncthreads();
    }
}

__global__ void gather3(const float* __restrict__ xyz, const int* __restrict__ idx,
                        int B_, int N, int S, float* __restrict__ out)
{
    int t = blockIdx.x * blockDim.x + threadIdx.x;
    if (t >= B_ * S * 3) return;
    int c = t % 3; int r = t / 3; int s = r % S; int b = r / S;
    out[t] = xyz[((size_t)b * N + idx[(size_t)b * S + s]) * 3 + c];
}

// Ball query: first K indices (ascending) within radius; pad with first hit.
__global__ void ball_query(const float* __restrict__ xyz, const float* __restrict__ new_xyz,
                           int B_, int N, int S, int K, float r2, int* __restrict__ out)
{
    int t = blockIdx.x * blockDim.x + threadIdx.x;
    if (t >= B_ * S) return;
    int b = t / S;
    const float* p = xyz + (size_t)b * N * 3;
    float cx = new_xyz[(size_t)t * 3 + 0];
    float cy = new_xyz[(size_t)t * 3 + 1];
    float cz = new_xyz[(size_t)t * 3 + 2];
    int* o = out + (size_t)t * K;
    int cnt = 0, first = -1;
    for (int i = 0; i < N && cnt < K; ++i) {
        float dx = p[i * 3] - cx, dy = p[i * 3 + 1] - cy, dz = p[i * 3 + 2] - cz;
        float d = dx * dx + dy * dy + dz * dz;
        if (d <= r2) { if (first < 0) first = i; o[cnt++] = i; }
    }
    if (first < 0) first = 0;
    for (; cnt < K; ++cnt) o[cnt] = first;
}

// Build grouped tensor gp[b,s,k, 0:3]=xyz[gi]-center, [3:3+Cf]=feat[gi],
// zero-padded out to Cpad channels (keeps GEMM rows 16B-aligned, K%32==0).
__global__ void group_kernel(const float* __restrict__ xyz, const float* __restrict__ feat,
                             int Cf, const float* __restrict__ new_xyz,
                             const int* __restrict__ gi,
                             int B_, int N, int S, int K, int Cpad,
                             float* __restrict__ out)
{
    size_t t = (size_t)blockIdx.x * blockDim.x + threadIdx.x;
    size_t total = (size_t)B_ * S * K * Cpad;
    if (t >= total) return;
    int c = (int)(t % Cpad);
    size_t r = t / Cpad;       // (b*S+j)*K + kk
    int kk = (int)(r % K);
    size_t cj = r / K;         // b*S + j
    int b = (int)(cj / S);
    float v = 0.f;
    if (c < 3 + Cf) {
        int src = gi[cj * K + kk];
        if (c < 3) v = xyz[((size_t)b * N + src) * 3 + c] - new_xyz[cj * 3 + c];
        else       v = feat[((size_t)b * N + src) * (size_t)Cf + (c - 3)];
    }
    out[t] = v;
}

// Zero-pad weight matrix [cout,cin] -> [cout,Cpad]
__global__ void pad_weights(const float* __restrict__ src, int cin, int cout, int Cpad,
                            float* __restrict__ dst)
{
    size_t t = (size_t)blockIdx.x * blockDim.x + threadIdx.x;
    size_t total = (size_t)cout * Cpad;
    if (t >= total) return;
    int c = (int)(t % Cpad);
    size_t r = t / Cpad;
    dst[t] = (c < cin) ? src[r * (size_t)cin + c] : 0.f;
}

// 3-NN inverse-distance interpolation of p2 onto xyz1, written @ col0 of out
__global__ void three_nn_interp(const float* __restrict__ xyz1, const float* __restrict__ xyz2,
                                const float* __restrict__ p2, int C2,
                                int B_, int N1, int N2,
                                float* __restrict__ out, int ldo, int col0)
{
    int t = blockIdx.x * blockDim.x + threadIdx.x;
    if (t >= B_ * N1) return;
    int b = t / N1;
    const float* q = xyz1 + (size_t)t * 3;
    const float* p = xyz2 + (size_t)b * N2 * 3;
    float d0 = 1e30f, d1 = 1e30f, d2 = 1e30f;
    int i0 = 0, i1 = 0, i2 = 0;
    for (int m = 0; m < N2; ++m) {
        float dx = p[m * 3] - q[0], dy = p[m * 3 + 1] - q[1], dz = p[m * 3 + 2] - q[2];
        float d = dx * dx + dy * dy + dz * dz;
        if (d < d0)      { d2 = d1; i2 = i1; d1 = d0; i1 = i0; d0 = d; i0 = m; }
        else if (d < d1) { d2 = d1; i2 = i1; d1 = d;  i1 = m; }
        else if (d < d2) { d2 = d;  i2 = m; }
    }
    float w0 = 1.f / (fmaxf(d0, 0.f) + 1e-8f);
    float w1 = 1.f / (fmaxf(d1, 0.f) + 1e-8f);
    float w2 = 1.f / (fmaxf(d2, 0.f) + 1e-8f);
    float ws = w0 + w1 + w2; w0 /= ws; w1 /= ws; w2 /= ws;
    const float* f = p2 + (size_t)b * N2 * C2;
    float* o = out + (size_t)t * ldo + col0;
    for (int c = 0; c < C2; ++c)
        o[c] = w0 * f[(size_t)i0 * C2 + c] + w1 * f[(size_t)i1 * C2 + c] + w2 * f[(size_t)i2 * C2 + c];
}

__global__ void copy_cols(const float* __restrict__ src, int lds_, int scol,
                          float* __restrict__ dst, int ldd, int dcol, int M, int C)
{
    size_t t = (size_t)blockIdx.x * blockDim.x + threadIdx.x;
    size_t total = (size_t)M * C;
    if (t >= total) return;
    int c = (int)(t % C);
    size_t m = t / C;
    dst[m * (size_t)ldd + dcol + c] = src[m * (size_t)lds_ + scol + c];
}

// nearest-neighbor channel-preserving upsample: dst[b,n,:] = src[b, n*S/N, :]
__global__ void nearest_up(const float* __restrict__ src, int S, int C, int B_, int N,
                           float* __restrict__ dst)
{
    size_t t = (size_t)blockIdx.x * blockDim.x + threadIdx.x;
    size_t total = (size_t)B_ * N * C;
    if (t >= total) return;
    int c = (int)(t % C);
    size_t r = t / C;
    int n = (int)(r % N);
    int b = (int)(r / N);
    int sidx = (int)(((long long)n * S) / N);
    dst[t] = src[(((size_t)b * S) + sidx) * (size_t)C + c];
}

// [B,N,C] -> [B,C,N]
__global__ void transpose_bcn(const float* __restrict__ src, float* __restrict__ dst,
                              int B_, int N, int C)
{
    size_t t = (size_t)blockIdx.x * blockDim.x + threadIdx.x;
    size_t total = (size_t)B_ * N * C;
    if (t >= total) return;
    int c = (int)(t % C);
    size_t r = t / C;
    int n = (int)(r % N);
    int b = (int)(r / N);
    dst[((size_t)b * C + c) * (size_t)N + n] = src[t];
}

// ---------------------------------------------------------------------------
// Host-side orchestration
// ---------------------------------------------------------------------------
struct Layer { const float *w, *b, *g, *be; int cin, cout; };

static inline int cdiv(int a, int b) { return (a + b - 1) / b; }
static inline unsigned gblocks(size_t total) { return (unsigned)((total + 255) / 256); }

static void run_gemm(hipStream_t st, const float* X, int M, int K,
                     const float* W, const float* bias, float* Y, int N)
{
    dim3 grid(cdiv(N, TN), cdiv(M, TM));
    gemm_bias_f16wmma<<<grid, 256, 0, st>>>(X, W, bias, Y, M, N, K);
}

static void bn_relu(hipStream_t st, float* Ytmp, int M, int C, float* stats,
                    const float* g, const float* be, const float* residual,
                    float* dst, int ldd, int dcol)
{
    bn_stats<<<C, 256, 0, st>>>(Ytmp, M, C, stats, stats + 2048);
    bn_apply_relu<<<gblocks((size_t)M * C), 256, 0, st>>>(
        Ytmp, M, C, stats, stats + 2048, g, be, residual, dst, ldd, dcol);
}

static void conv_bn_relu(hipStream_t st, const float* X, int M, const Layer& L,
                         float* Ytmp, float* stats,
                         const float* residual, float* dst, int ldd, int dcol)
{
    run_gemm(st, X, M, L.cin, L.w, L.b, Ytmp, L.cout);
    bn_relu(st, Ytmp, M, L.cout, stats, L.g, L.be, residual, dst, ldd, dcol);
}

extern "C" void kernel_launch(void* const* d_in, const int* in_sizes, int n_in,
                              void* d_out, int out_size, void* d_ws, size_t ws_size,
                              hipStream_t stream)
{
    (void)in_sizes; (void)n_in; (void)out_size; (void)ws_size;
    const int B = 4, N0 = 16384;
    const float* xyz      = (const float*)d_in[0];
    const float* features = (const float*)d_in[1];

    // Params flattened as JAX pytree: top-level keys sorted, each _mk dict
    // yields leaves in order b, be, g, w.
    auto get = [&](int base, int cin, int cout) {
        Layer L;
        L.b  = (const float*)d_in[base + 0];
        L.be = (const float*)d_in[base + 1];
        L.g  = (const float*)d_in[base + 2];
        L.w  = (const float*)d_in[base + 3];
        L.cin = cin; L.cout = cout; return L;
    };
    Layer bri0 = get(2, 3, 32),  bri1 = get(6, 32, 3);
    Layer color0 = get(10, 3, 6);
    Layer final1 = get(14, 384, 128), final2 = get(18, 128, 5);
    Layer fp1_0 = get(22, 256, 256),  fp1_1 = get(26, 256, 128);
    Layer fp2_0 = get(30, 512, 256),  fp2_1 = get(34, 256, 256);
    Layer fp3_0 = get(38, 1536, 1024), fp3_1 = get(42, 1024, 256);
    Layer fuse0 = get(46, 9, 3);
    Layer fus_l2 = get(50, 256, 128), fus_l1 = get(54, 256, 128), fus_l0 = get(58, 128, 128);
    Layer geo2 = get(62, 512, 512), geo3 = get(66, 1024, 1024);
    Layer sa1m0[3] = { get(70, 6, 64),   get(74, 64, 64),   get(78, 64, 128) };
    Layer sa1m1[3] = { get(82, 6, 64),   get(86, 64, 64),   get(90, 64, 128) };
    Layer sa2m0[3] = { get(94, 259, 128), get(98, 128, 128), get(102, 128, 256) };
    Layer sa2m1[3] = { get(106, 259, 128), get(110, 128, 128), get(114, 128, 256) };
    Layer sa3m0[3] = { get(118, 515, 256), get(122, 256, 256), get(126, 256, 512) };
    Layer sa3m1[3] = { get(130, 515, 256), get(134, 256, 256), get(138, 256, 512) };

    // ---- workspace carve (deterministic, recomputed every call) ----
    char* wp = (char*)d_ws;
    auto af = [&](size_t n) { float* r = (float*)wp; wp += ((n * sizeof(float) + 255) & ~(size_t)255); return r; };
    auto ai = [&](size_t n) { int*   r = (int*)wp;   wp += ((n * sizeof(int)   + 255) & ~(size_t)255); return r; };

    float* stats = af(4096);
    float* Wpad  = af((size_t)256 * 544);         // padded layer-0 weights
    float* dist  = af((size_t)B * N0);
    int*   fidx  = ai((size_t)B * 1024);
    int*   gi    = ai((size_t)B * 1024 * 32);
    float* pe    = af((size_t)B * N0 * 3);
    float* colf  = af((size_t)B * N0 * 6);
    float* cat9  = af((size_t)B * N0 * 9);
    float* fused = af((size_t)B * N0 * 3);
    float* l1x = af((size_t)B * 1024 * 3);
    float* l2x = af((size_t)B * 512 * 3);
    float* l3x = af((size_t)B * 128 * 3);
    float* l1f  = af((size_t)B * 1024 * 256);
    float* l2f  = af((size_t)B * 512 * 512);
    float* l3f  = af((size_t)B * 128 * 1024);
    float* l2f2 = af((size_t)B * 512 * 256);
    float* l1f2 = af((size_t)B * 1024 * 256);
    float* l0f  = af((size_t)B * N0 * 128);
    float* fus  = af((size_t)B * N0 * 384);
    float* Xb   = af((size_t)20 * 1024 * 1024);   // grouped tensors / FP concat inputs
    float* T0   = af((size_t)17 * 1024 * 1024);   // ping
    float* T1   = af((size_t)17 * 1024 * 1024);   // pong
    float* outT = af((size_t)B * N0 * 5);

    const int Mfull = B * N0;

    // ---- stage 0: bri / color / fuse ----
    conv_bn_relu(stream, xyz,      Mfull, bri0,  T0, stats, nullptr, T0, 32, 0);
    conv_bn_relu(stream, T0,       Mfull, bri1,  T1, stats, nullptr, pe,  3, 0);
    conv_bn_relu(stream, features, Mfull, color0, T0, stats, nullptr, colf, 6, 0);
    copy_cols<<<gblocks((size_t)Mfull * 3), 256, 0, stream>>>(pe,   3, 0, cat9, 9, 0, Mfull, 3);
    copy_cols<<<gblocks((size_t)Mfull * 6), 256, 0, stream>>>(colf, 6, 0, cat9, 9, 3, Mfull, 6);
    conv_bn_relu(stream, cat9, Mfull, fuse0, T0, stats, nullptr, fused, 3, 0);

    // ---- multi-scale set-abstraction stage ----
    auto sa_stage = [&](const float* in_xyz, int Sin, const float* in_feat, int Cf,
                        int Sout, float r0, int K0, const Layer* m0,
                        float r1, int K1, const Layer* m1,
                        float* out_xyz, float* out_feat, int Ctot)
    {
        fps_kernel<<<B, 256, 0, stream>>>(in_xyz, B, Sin, Sout, fidx, dist);
        gather3<<<gblocks((size_t)B * Sout * 3), 256, 0, stream>>>(in_xyz, fidx, B, Sin, Sout, out_xyz);
        const float rr[2] = { r0, r1 };
        const int   kk[2] = { K0, K1 };
        const Layer* ms[2] = { m0, m1 };
        int dcol = 0;
        for (int s = 0; s < 2; ++s) {
            const int K = kk[s];
            ball_query<<<gblocks((size_t)B * Sout), 128, 0, stream>>>(
                in_xyz, out_xyz, B, Sin, Sout, K, rr[s] * rr[s], gi);
            const int Cpad = (3 + Cf + 31) & ~31;   // grouped input padded to k%32==0
            size_t tot = (size_t)B * Sout * K * Cpad;
            group_kernel<<<gblocks(tot), 256, 0, stream>>>(
                in_xyz, in_feat, Cf, out_xyz, gi, B, Sin, Sout, K, Cpad, Xb);
            const int M = B * Sout * K;
            // layer 0 with zero-padded weights (aligned fast-path staging)
            pad_weights<<<gblocks((size_t)ms[s][0].cout * Cpad), 256, 0, stream>>>(
                ms[s][0].w, ms[s][0].cin, ms[s][0].cout, Cpad, Wpad);
            run_gemm(stream, Xb, M, Cpad, Wpad, ms[s][0].b, T0, ms[s][0].cout);
            bn_relu(stream, T0, M, ms[s][0].cout, stats, ms[s][0].g, ms[s][0].be,
                    nullptr, T0, ms[s][0].cout, 0);
            conv_bn_relu(stream, T0, M, ms[s][1], T1, stats, nullptr, T1, ms[s][1].cout, 0);
            conv_bn_relu(stream, T1, M, ms[s][2], T0, stats, nullptr, T0, ms[s][2].cout, 0);
            size_t pt = (size_t)B * Sout * ms[s][2].cout;
            maxpool_over_k<<<gblocks(pt), 256, 0, stream>>>(
                T0, K, ms[s][2].cout, B * Sout, out_feat, Ctot, dcol);
            dcol += ms[s][2].cout;
        }
    };

    sa_stage(xyz, N0,  fused, 3,   1024, 0.1f, 16, sa1m0, 0.2f, 32, sa1m1, l1x, l1f, 256);
    sa_stage(l1x, 1024, l1f, 256,  512,  0.2f, 16, sa2m0, 0.4f, 32, sa2m1, l2x, l2f, 512);
    conv_bn_relu(stream, l2f, B * 512, geo2, T0, stats, l2f, l2f, 512, 0);       // _geo residual
    sa_stage(l2x, 512,  l2f, 512,  128,  0.4f, 16, sa3m0, 0.8f, 32, sa3m1, l3x, l3f, 1024);
    conv_bn_relu(stream, l3f, B * 128, geo3, T0, stats, l3f, l3f, 1024, 0);      // _geo residual

    // ---- feature propagation ----
    auto fp_stage = [&](const float* x1, int N1, const float* x2, int N2,
                        const float* p1, int C1, const float* p2, int C2,
                        const Layer& La, const Layer& Lb, float* out)
    {
        const int Cin = C1 + C2;
        const int M = B * N1;
        if (p1)
            copy_cols<<<gblocks((size_t)M * C1), 256, 0, stream>>>(p1, C1, 0, Xb, Cin, 0, M, C1);
        three_nn_interp<<<gblocks((size_t)B * N1), 128, 0, stream>>>(
            x1, x2, p2, C2, B, N1, N2, Xb, Cin, C1);
        conv_bn_relu(stream, Xb, M, La, T0, stats, nullptr, T0, La.cout, 0);
        conv_bn_relu(stream, T0, M, Lb, T1, stats, nullptr, out, Lb.cout, 0);
    };

    fp_stage(l2x, 512,  l3x, 128,  l2f, 512, l3f,  1024, fp3_0, fp3_1, l2f2);
    fp_stage(l1x, 1024, l2x, 512,  l1f, 256, l2f2, 256,  fp2_0, fp2_1, l1f2);
    fp_stage(xyz, N0,   l1x, 1024, nullptr, 0, l1f2, 256, fp1_0, fp1_1, l0f);

    // ---- fusion heads (nearest upsample + conv, written into fus columns) ----
    auto fusion_branch = [&](const float* f, int S, int C, const Layer& L, int dcol) {
        size_t tot = (size_t)B * N0 * C;
        nearest_up<<<gblocks(tot), 256, 0, stream>>>(f, S, C, B, N0, T0);
        conv_bn_relu(stream, T0, B * N0, L, T1, stats, nullptr, fus, 384, dcol);
    };
    fusion_branch(l2f2, 512,  256, fus_l2, 0);
    fusion_branch(l1f2, 1024, 256, fus_l1, 128);
    fusion_branch(l0f,  N0,   128, fus_l0, 256);

    // ---- final layers ----
    conv_bn_relu(stream, fus, Mfull, final1, T0, stats, nullptr, T0, 128, 0);
    run_gemm(stream, T0, Mfull, 128, final2.w, final2.b, outT, 5);
    transpose_bcn<<<gblocks((size_t)B * N0 * 5), 256, 0, stream>>>(outT, (float*)d_out, B, N0, 5);
}
